// EquivariantToyResFlow_25237227831587
// MI455X (gfx1250) — matrix operand order repaired and checked by reading, not compiled, not run on hardware
//
#include <hip/hip_runtime.h>
#include <hip/hip_bf16.h>

// ---------------------------------------------------------------------------
// EquivariantToyResFlow on MI455X (gfx1250), fully fused per 64-row tile.
// Mid layers (256x256) run on v_wmma_f32_16x16x32_bf16; activations stay in
// LDS for all 8 blocks x (1 primal + 5 JVP) passes. Weights transposed to
// bf16 once per launch into d_ws.
// Each wave owns TWO 16-row M-tiles so every B-fragment load feeds 2 WMMAs
// (halves WGP$/L2 weight traffic per FLOP).
// ---------------------------------------------------------------------------

typedef __attribute__((ext_vector_type(16))) __bf16 v16bf;
typedef __attribute__((ext_vector_type(8)))  __bf16 bf16x8;
typedef __attribute__((ext_vector_type(8)))  float  v8f;

#define BATCH  131072
#define HDIM   256
#define NBLK   8
#define NMID   3
#define NPS    5
#define MT     64      // rows per workgroup
#define XPITCH 264     // padded activation row pitch (bf16 elems), 528B = 33*16B

struct SMem {
  __bf16 bufA[MT][XPITCH];            // activation ping
  __bf16 bufB[MT][XPITCH];            // activation pong
  unsigned char msk[4][MT][HDIM];     // relu masks: after in-layer + 3 mids
  float wiL[2][HDIM];
  float biL[HDIM];
  float bmL[NMID][HDIM];
  float woL[HDIM][2];
  float boL[2];
  float xz[MT][2];                    // running z
  float vv[MT][2];                    // Hutchinson probe
  float gx[MT][2];                    // g(x) for current block
  float tw[MT][2];                    // current tangent (2-dim ends)
  float dlp[MT];                      // delta_logp accumulator
  unsigned char msk4[MT][2];          // relu mask of output layer
};

// A-fragment (16x32 bf16, M x K): lane = {m = L%16, half = L/16};
// elements 0..7 -> K = ks*32 + half*8 + (0..7); 8..15 -> +16.
__device__ __forceinline__ v16bf load_a_frag(const __bf16* row, int ks, int half) {
  const bf16x8 lo = *(const bf16x8*)(row + ks * 32 + half * 8);
  const bf16x8 hi = *(const bf16x8*)(row + ks * 32 + 16 + half * 8);
  v16bf r;
#pragma unroll
  for (int i = 0; i < 8; ++i) { r[i] = lo[i]; r[i + 8] = hi[i]; }
  return r;
}

// B-fragment (32x16 bf16, K x N): lane = {n = L%16, half = L/16};
// elements 0..7 -> K = ks*32 + half*8 + (0..7); 8..15 -> +16. wrow = &WmT[n][0] (K-contig).
__device__ __forceinline__ v16bf load_b_frag(const __bf16* wrow, int ks, int half) {
  const bf16x8 lo = *(const bf16x8*)(wrow + ks * 32 + half * 8);
  const bf16x8 hi = *(const bf16x8*)(wrow + ks * 32 + 16 + half * 8);
  v16bf r;
#pragma unroll
  for (int i = 0; i < 8; ++i) { r[i] = lo[i]; r[i + 8] = hi[i]; }
  return r;
}

// One 64x256 @ 256x256 layer. PRIMAL: +bias, relu, write mask.
// Tangent: multiply by stored mask. All 8 waves participate (EXEC all ones).
// Wave mapping: mg = wid&1 owns rows [mg*32, mg*32+32) (two 16-row M-tiles),
// ng = wid>>1 owns N-tiles [ng*4, ng*4+4). Each B-fragment is used by 2 WMMAs.
template <bool PRIMAL>
__device__ __forceinline__ void mid_layer(const __bf16 (*X)[XPITCH],
                                          __bf16 (*Y)[XPITCH],
                                          unsigned char (*mk)[HDIM],
                                          const __bf16* __restrict__ Wl,
                                          const float* bias, int tid) {
  const int wid = tid >> 5, lane = tid & 31;
  const int mg = wid & 1, ng = wid >> 1;
  const int lm = lane & 15, half = lane >> 4;
  const int mrow0 = mg * 32 + lm;       // M-tile 2*mg
  const int mrow1 = mrow0 + 16;         // M-tile 2*mg+1

  v16bf a0[8], a1[8];
#pragma unroll
  for (int ks = 0; ks < 8; ++ks) {
    a0[ks] = load_a_frag(&X[mrow0][0], ks, half);
    a1[ks] = load_a_frag(&X[mrow1][0], ks, half);
  }

  for (int t = 0; t < 4; ++t) {
    const int nt = ng * 4 + t;
    const int n  = nt * 16 + lm;
    const __bf16* wrow = Wl + (size_t)n * HDIM;
    v8f acc0 = {0.f, 0.f, 0.f, 0.f, 0.f, 0.f, 0.f, 0.f};
    v8f acc1 = {0.f, 0.f, 0.f, 0.f, 0.f, 0.f, 0.f, 0.f};
#pragma unroll
    for (int ks = 0; ks < 8; ++ks) {
      const v16bf bfrag = load_b_frag(wrow, ks, half);
      acc0 = __builtin_amdgcn_wmma_f32_16x16x32_bf16(
          false, a0[ks], false, bfrag, (short)0, acc0, false, false);
      acc1 = __builtin_amdgcn_wmma_f32_16x16x32_bf16(
          false, a1[ks], false, bfrag, (short)0, acc1, false, false);
    }
    if (PRIMAL) {
      const float bb = bias[n];
#pragma unroll
      for (int j = 0; j < 8; ++j) {
        // C layout: VGPR j holds row j (lanes 0-15) / j+8 (lanes 16-31)
        const int mm0 = mg * 32 + j + 8 * half;
        const int mm1 = mm0 + 16;
        const float pre0 = acc0[j] + bb;
        const float pre1 = acc1[j] + bb;
        const bool pos0 = pre0 > 0.f;
        const bool pos1 = pre1 > 0.f;
        Y[mm0][n] = (__bf16)(pos0 ? pre0 : 0.f);
        Y[mm1][n] = (__bf16)(pos1 ? pre1 : 0.f);
        mk[mm0][n] = pos0 ? (unsigned char)1 : (unsigned char)0;
        mk[mm1][n] = pos1 ? (unsigned char)1 : (unsigned char)0;
      }
    } else {
#pragma unroll
      for (int j = 0; j < 8; ++j) {
        const int mm0 = mg * 32 + j + 8 * half;
        const int mm1 = mm0 + 16;
        Y[mm0][n] = (__bf16)(mk[mm0][n] ? acc0[j] : 0.f);
        Y[mm1][n] = (__bf16)(mk[mm1][n] ? acc1[j] : 0.f);
      }
    }
  }
}

// Transpose + convert Wm[b][j][k][n] (f32) -> WmT[b][j][n][k] (bf16).
__global__ void resflow_prep_kernel(const float* __restrict__ Wm,
                                    __bf16* __restrict__ WmT) {
  const int idx = blockIdx.x * blockDim.x + threadIdx.x;
  const int total = NBLK * NMID * HDIM * HDIM;
  if (idx >= total) return;
  const int k = idx & 255;
  const int n = (idx >> 8) & 255;
  const int l = idx >> 16;                  // l = b*NMID + j
  WmT[idx] = (__bf16)Wm[((size_t)l * HDIM + k) * HDIM + n];
}

__global__ __launch_bounds__(256)
void resflow_kernel(const float* __restrict__ x, const float* __restrict__ v,
                    const float* __restrict__ Wi, const float* __restrict__ bi,
                    const float* __restrict__ bm, const float* __restrict__ Wo,
                    const float* __restrict__ bo,
                    const __bf16* __restrict__ WmT,
                    float* __restrict__ out) {
  __shared__ SMem sm;
  const int tid = threadIdx.x;
  const int row0 = blockIdx.x * MT;

  if (tid < 128) {
    const int m = tid >> 1, c = tid & 1;
    sm.xz[m][c] = x[(size_t)(row0 + m) * 2 + c];
    sm.vv[m][c] = v[(size_t)(row0 + m) * 2 + c];
  }
  if (tid < MT) sm.dlp[tid] = 0.f;
  __syncthreads();

  for (int b = 0; b < NBLK; ++b) {
    // ---- stage per-block params into LDS ----
    for (int i = tid; i < 2 * HDIM; i += 256)
      ((float*)sm.wiL)[i] = Wi[(size_t)b * 2 * HDIM + i];
    for (int i = tid; i < HDIM; i += 256)
      sm.biL[i] = bi[(size_t)b * HDIM + i];
    for (int i = tid; i < NMID * HDIM; i += 256)
      ((float*)sm.bmL)[i] = bm[(size_t)b * NMID * HDIM + i];
    for (int i = tid; i < HDIM * 2; i += 256)
      ((float*)sm.woL)[i] = Wo[(size_t)b * HDIM * 2 + i];
    if (tid < 2) sm.boL[tid] = bo[b * 2 + tid];
    __syncthreads();

    const __bf16* Wb = WmT + (size_t)b * NMID * HDIM * HDIM;

    // ---- primal pass ----
    {  // input layer: one thread per output column
      const int n = tid;
      const float w0 = sm.wiL[0][n], w1 = sm.wiL[1][n], bb = sm.biL[n];
      for (int m = 0; m < MT; ++m) {
        const float pre = sm.xz[m][0] * w0 + sm.xz[m][1] * w1 + bb;
        const bool pos = pre > 0.f;
        sm.bufA[m][n] = (__bf16)(pos ? pre : 0.f);
        sm.msk[0][m][n] = pos ? (unsigned char)1 : (unsigned char)0;
      }
    }
    __syncthreads();
    mid_layer<true>(sm.bufA, sm.bufB, sm.msk[1], Wb + 0 * HDIM * HDIM, sm.bmL[0], tid);
    __syncthreads();
    mid_layer<true>(sm.bufB, sm.bufA, sm.msk[2], Wb + 1 * HDIM * HDIM, sm.bmL[1], tid);
    __syncthreads();
    mid_layer<true>(sm.bufA, sm.bufB, sm.msk[3], Wb + 2 * HDIM * HDIM, sm.bmL[2], tid);
    __syncthreads();
    if (tid < 128) {  // output layer (h3 in bufB); also init tangent w = v
      const int m = tid >> 1, c = tid & 1;
      float acc = 0.f;
      for (int k = 0; k < HDIM; ++k) acc += (float)sm.bufB[m][k] * sm.woL[k][c];
      const float pre = acc + sm.boL[c];
      const bool pos = pre > 0.f;
      sm.gx[m][c] = pos ? pre : 0.f;
      sm.msk4[m][c] = pos ? (unsigned char)1 : (unsigned char)0;
      sm.tw[m][c] = sm.vv[m][c];
    }
    __syncthreads();

    // ---- power-series JVP passes ----
    for (int p = 1; p <= NPS; ++p) {
      const float coef = ((p & 1) ? 1.f : -1.f) / (float)p;
      {  // tangent input layer
        const int n = tid;
        const float w0 = sm.wiL[0][n], w1 = sm.wiL[1][n];
        for (int m = 0; m < MT; ++m) {
          const float pre = sm.tw[m][0] * w0 + sm.tw[m][1] * w1;
          sm.bufA[m][n] = (__bf16)(sm.msk[0][m][n] ? pre : 0.f);
        }
      }
      __syncthreads();
      mid_layer<false>(sm.bufA, sm.bufB, sm.msk[1], Wb + 0 * HDIM * HDIM, nullptr, tid);
      __syncthreads();
      mid_layer<false>(sm.bufB, sm.bufA, sm.msk[2], Wb + 1 * HDIM * HDIM, nullptr, tid);
      __syncthreads();
      mid_layer<false>(sm.bufA, sm.bufB, sm.msk[3], Wb + 2 * HDIM * HDIM, nullptr, tid);
      __syncthreads();
      if (tid < 128) {  // tangent output layer + log-det contribution
        const int m = tid >> 1, c = tid & 1;
        float acc = 0.f;
        for (int k = 0; k < HDIM; ++k) acc += (float)sm.bufB[m][k] * sm.woL[k][c];
        const float s = sm.msk4[m][c] ? acc : 0.f;
        sm.tw[m][c] = s;  // w <- J_g w  (input-layer reads of tw are long done)
        atomicAdd(&sm.dlp[m], -coef * s * sm.vv[m][c]);  // delta_logp -= coef*<w,v>
      }
      __syncthreads();
    }

    // ---- residual update: z <- z + g(z) ----
    if (tid < 128) {
      const int m = tid >> 1, c = tid & 1;
      sm.xz[m][c] += sm.gx[m][c];
    }
    __syncthreads();
  }

  if (tid < 128) {
    const int m = tid >> 1, c = tid & 1;
    out[(size_t)(row0 + m) * 2 + c] = sm.xz[m][c];
  }
  if (tid < MT) out[(size_t)BATCH * 2 + row0 + tid] = sm.dlp[tid];
}

extern "C" void kernel_launch(void* const* d_in, const int* in_sizes, int n_in,
                              void* d_out, int out_size, void* d_ws, size_t ws_size,
                              hipStream_t stream) {
  (void)in_sizes; (void)n_in; (void)out_size; (void)ws_size;
  const float* x  = (const float*)d_in[0];
  const float* v  = (const float*)d_in[1];
  const float* Wi = (const float*)d_in[2];
  const float* bi = (const float*)d_in[3];
  const float* Wm = (const float*)d_in[4];
  const float* bm = (const float*)d_in[5];
  const float* Wo = (const float*)d_in[6];
  const float* bo = (const float*)d_in[7];

  __bf16* WmT = (__bf16*)d_ws;  // NBLK*NMID*HDIM*HDIM bf16 = 3 MB scratch

  const int totalW = NBLK * NMID * HDIM * HDIM;
  resflow_prep_kernel<<<(totalW + 255) / 256, 256, 0, stream>>>(Wm, WmT);
  resflow_kernel<<<BATCH / MT, 256, 0, stream>>>(x, v, Wi, bi, bm, Wo, bo, WmT,
                                                 (float*)d_out);
}
